// GRUDecoder_35605278884065
// MI455X (gfx1250) — compile-verified, matrix operand order checked by reading, not run on hardware
//
#include <hip/hip_runtime.h>

// ---------------------------------------------------------------------------
// GRU decoder with Bahdanau attention for MI455X (gfx1250, wave32, WMMA).
// B=64, T=64, S=128, E=512, H=1024, V=32000.
// All big GEMMs run as v_wmma_f32_16x16x32_bf16 with fp32 accumulation;
// weights/activations converted to bf16 once in workspace (L2-resident).
// ---------------------------------------------------------------------------

typedef __attribute__((ext_vector_type(8)))  __bf16 v8bf;
typedef __attribute__((ext_vector_type(16))) __bf16 v16bf;
typedef __attribute__((ext_vector_type(8)))  float  v8f;

#define DB 64
#define DT 64
#define DS 128
#define DE 512
#define DH 1024

// Load one 16x32 bf16 WMMA fragment row for this lane.
// Layout (CDNA5 ISA 7.12.2, 16-bit A 16x32):
//   lane = row + 16*kg ; elements 0..7 -> K = k0 + kg*8 + (0..7)
//                         elements 8..15 -> K = k0 + 16 + kg*8 + (0..7)
// B fragments use the identical pattern with "row" = output column (NT gemm).
__device__ __forceinline__ v16bf load_frag(const __bf16* __restrict__ rowbase,
                                           int k0, int kg) {
  const __bf16* p = rowbase + k0 + kg * 8;
  v8bf lo = *(const v8bf*)(p);
  v8bf hi = *(const v8bf*)(p + 16);
  return __builtin_shufflevector(lo, hi, 0, 1, 2, 3, 4, 5, 6, 7,
                                 8, 9, 10, 11, 12, 13, 14, 15);
}

// C[M,N] = A[M,K](row-major) * Bw[N,K](row-major, dot over K) + bias
// One wave -> 16x64 tile (4 wmma accumulators sharing one A fragment).
__global__ void __launch_bounds__(256)
wmma_gemm_nt(const __bf16* __restrict__ A, int lda,
             const __bf16* __restrict__ Bw, int ldb,
             float* __restrict__ C, int ldc,
             const float* __restrict__ bias,
             int tiles_n, int K, int tiles_total) {
  const int wave = threadIdx.x >> 5;
  const int lane = threadIdx.x & 31;
  const int tile = blockIdx.x * 8 + wave;
  if (tile >= tiles_total) return;
  const int tm = tile / tiles_n;
  const int tn = tile % tiles_n;
  const int row = lane & 15;
  const int kg  = lane >> 4;

  const __bf16* Arow = A  + (size_t)(tm * 16 + row) * lda;
  const __bf16* Brow = Bw + (size_t)(tn * 64 + row) * ldb;
  const size_t bstep = (size_t)16 * ldb;

  v8f acc[4];
  for (int i = 0; i < 4; ++i) acc[i] = (v8f)0.0f;

  for (int k0 = 0; k0 < K; k0 += 32) {
    if (k0 + 32 < K) {
      __builtin_prefetch(Arow + k0 + 32, 0, 1);
      __builtin_prefetch(Brow + k0 + 32, 0, 1);
    }
    v16bf a  = load_frag(Arow, k0, kg);
    v16bf b0 = load_frag(Brow,             k0, kg);
    v16bf b1 = load_frag(Brow + bstep,     k0, kg);
    v16bf b2 = load_frag(Brow + 2 * bstep, k0, kg);
    v16bf b3 = load_frag(Brow + 3 * bstep, k0, kg);
    acc[0] = __builtin_amdgcn_wmma_f32_16x16x32_bf16(false, a, false, b0,
                                                     (short)0, acc[0], false, false);
    acc[1] = __builtin_amdgcn_wmma_f32_16x16x32_bf16(false, a, false, b1,
                                                     (short)0, acc[1], false, false);
    acc[2] = __builtin_amdgcn_wmma_f32_16x16x32_bf16(false, a, false, b2,
                                                     (short)0, acc[2], false, false);
    acc[3] = __builtin_amdgcn_wmma_f32_16x16x32_bf16(false, a, false, b3,
                                                     (short)0, acc[3], false, false);
  }

  // C/D layout: VGPR e holds M = tm*16 + e + 8*kg, N = tn*64 + jt*16 + (lane&15)
  const int ncol = lane & 15;
  for (int jt = 0; jt < 4; ++jt) {
    const int n = tn * 64 + jt * 16 + ncol;
    const float bv = bias ? bias[n] : 0.0f;
    for (int e = 0; e < 8; ++e) {
      const int m = tm * 16 + e + 8 * kg;
      C[(size_t)m * ldc + n] = acc[jt][e] + bv;
    }
  }
}

// ---- fp32 -> bf16 flat convert -------------------------------------------
__global__ void cvt_bf16(const float* __restrict__ in, __bf16* __restrict__ out,
                         size_t n) {
  size_t stride = (size_t)gridDim.x * blockDim.x;
  for (size_t i = blockIdx.x * (size_t)blockDim.x + threadIdx.x; i < n; i += stride)
    out[i] = (__bf16)in[i];
}

// in: (Krows x Ncols) fp32 row-major ; out: (Ncols x Krows) bf16 row-major
__global__ void cvt_bf16_t(const float* __restrict__ in, __bf16* __restrict__ out,
                           int Krows, int Ncols) {
  size_t n = (size_t)Krows * Ncols;
  size_t stride = (size_t)gridDim.x * blockDim.x;
  for (size_t i = blockIdx.x * (size_t)blockDim.x + threadIdx.x; i < n; i += stride) {
    int nn = (int)(i / Krows);
    int kk = (int)(i % Krows);
    out[i] = (__bf16)in[(size_t)kk * Ncols + nn];
  }
}

// ---- h0 = tanh(encoder_final[-1] @ bridge_W + bridge_b) ------------------
__global__ void __launch_bounds__(256)
h0_kernel(const float* __restrict__ efin, const float* __restrict__ bw,
          const float* __restrict__ bb, float* __restrict__ h,
          __bf16* __restrict__ hbf) {
  const int b = blockIdx.x;
  const float* e = efin + (size_t)b * (2 * DH);
  for (int j = threadIdx.x; j < DH; j += 256) {
    float s = bb[j];
    for (int k = 0; k < 2 * DH; ++k)
      s = fmaf(e[k], bw[(size_t)k * DH + j], s);
    float v = tanhf(s);
    h[(size_t)b * DH + j]   = v;
    hbf[(size_t)b * DH + j] = (__bf16)v;
  }
}

// ---- embedding gather for step t -> bf16 into xbuf[:, :E], cat2[:, :E] ---
__global__ void embed_kernel(const int* __restrict__ trg,
                             const float* __restrict__ tbl,
                             __bf16* __restrict__ xbuf,
                             __bf16* __restrict__ cat2, int t) {
  const int b = blockIdx.x;
  const int tok = trg[b * DT + t];
  for (int e = threadIdx.x; e < DE; e += 128) {
    __bf16 v = (__bf16)tbl[(size_t)tok * DE + e];
    xbuf[(size_t)b * (DE + 2 * DH) + e] = v;
    cat2[(size_t)b * (DE + 3 * DH) + e] = v;
  }
}

// ---- attention: scores -> masked softmax -> context (one block per batch)
__global__ void __launch_bounds__(256)
attn_kernel(const float* __restrict__ h, const float* __restrict__ pk,
            const float* __restrict__ ehid, const unsigned char* __restrict__ mask,
            __bf16* __restrict__ xbuf, __bf16* __restrict__ cat2) {
  const int b = blockIdx.x;
  const int t = threadIdx.x;
  __shared__ float hs[DH];
  __shared__ float sc[DS];
  __shared__ float smax, sinv;

  for (int i = t; i < DH; i += 256) hs[i] = h[(size_t)b * DH + i];
  __syncthreads();

  if (t < DS) {
    const float* pkr = pk + ((size_t)b * DS + t) * DH;
    float s = 0.0f;
    for (int k = 0; k < DH; ++k) s = fmaf(hs[k], pkr[k], s);
    sc[t] = mask[b * DS + t] ? s : -1.0e9f;
  }
  __syncthreads();
  if (t == 0) {
    float m = sc[0];
    for (int i = 1; i < DS; ++i) m = fmaxf(m, sc[i]);
    smax = m;
  }
  __syncthreads();
  if (t < DS) sc[t] = __expf(sc[t] - smax);
  __syncthreads();
  if (t == 0) {
    float s = 0.0f;
    for (int i = 0; i < DS; ++i) s += sc[i];
    sinv = 1.0f / s;
  }
  __syncthreads();
  if (t < DS) sc[t] *= sinv;
  __syncthreads();

  for (int d = t; d < 2 * DH; d += 256) {
    const float* er = ehid + (size_t)b * DS * (2 * DH) + d;
    float c = 0.0f;
    for (int s = 0; s < DS; ++s)
      c = fmaf(sc[s], er[(size_t)s * (2 * DH)], c);
    __bf16 cb = (__bf16)c;
    xbuf[(size_t)b * (DE + 2 * DH) + DE + d]      = cb;
    cat2[(size_t)b * (DE + 3 * DH) + DE + DH + d] = cb;
  }
}

// ---- GRU gate pointwise ---------------------------------------------------
__global__ void __launch_bounds__(256)
gate_kernel(const float* __restrict__ gx, const float* __restrict__ gh,
            const float* __restrict__ bih, const float* __restrict__ bhh,
            float* __restrict__ h, __bf16* __restrict__ hbf,
            __bf16* __restrict__ cat2) {
  const int b = blockIdx.x;
  const size_t base = (size_t)b * (3 * DH);
  for (int j = threadIdx.x; j < DH; j += 256) {
    float xr = gx[base + j]          + bih[j];
    float xz = gx[base + DH + j]     + bih[DH + j];
    float xn = gx[base + 2 * DH + j] + bih[2 * DH + j];
    float hr = gh[base + j]          + bhh[j];
    float hz = gh[base + DH + j]     + bhh[DH + j];
    float hn = gh[base + 2 * DH + j] + bhh[2 * DH + j];
    float r = 1.0f / (1.0f + __expf(-(xr + hr)));
    float z = 1.0f / (1.0f + __expf(-(xz + hz)));
    float n = tanhf(xn + r * hn);
    float hv = h[(size_t)b * DH + j];
    float hnew = (1.0f - z) * n + z * hv;
    h[(size_t)b * DH + j]   = hnew;
    hbf[(size_t)b * DH + j] = (__bf16)hnew;
    cat2[(size_t)b * (DE + 3 * DH) + DE + j] = (__bf16)hnew;
  }
}

// ---------------------------------------------------------------------------
extern "C" void kernel_launch(void* const* d_in, const int* in_sizes, int n_in,
                              void* d_out, int out_size, void* d_ws, size_t ws_size,
                              hipStream_t stream) {
  (void)in_sizes; (void)n_in; (void)out_size; (void)ws_size;
  const int*           trg  = (const int*)d_in[0];
  const float*         ehid = (const float*)d_in[1];
  const unsigned char* mask = (const unsigned char*)d_in[2];
  /* d_in[3] tgt_mask unused */
  const float* efin = (const float*)d_in[4];
  const float* etbl = (const float*)d_in[5];
  const float* Wk   = (const float*)d_in[6];
  const float* Wih  = (const float*)d_in[7];
  const float* Whh  = (const float*)d_in[8];
  const float* bih  = (const float*)d_in[9];
  const float* bhh  = (const float*)d_in[10];
  const float* brW  = (const float*)d_in[11];
  const float* brb  = (const float*)d_in[12];
  const float* Wp   = (const float*)d_in[13];
  const float* bp   = (const float*)d_in[14];
  float* out = (float*)d_out;

  // ---- workspace carve (aligned 256B) ----
  char* ws = (char*)d_ws;
  size_t off = 0;
  auto alloc = [&](size_t bytes) -> void* {
    off = (off + 255) & ~(size_t)255;
    void* p = ws + off;
    off += bytes;
    return p;
  };
  __bf16* ehid_bf = (__bf16*)alloc((size_t)DB * DS * 2 * DH * 2);   // 33.5 MB
  float*  pk      = (float*) alloc((size_t)DB * DS * DH * 4);       // 33.5 MB
  __bf16* Wih_bf  = (__bf16*)alloc((size_t)3 * DH * (DE + 2 * DH) * 2);
  __bf16* Whh_bf  = (__bf16*)alloc((size_t)3 * DH * DH * 2);
  __bf16* WpT_bf  = (__bf16*)alloc((size_t)DH * (DE + 3 * DH) * 2);
  __bf16* WkT_bf  = (__bf16*)alloc((size_t)DH * 2 * DH * 2);
  __bf16* xbuf    = (__bf16*)alloc((size_t)DB * (DE + 2 * DH) * 2);
  __bf16* cat2    = (__bf16*)alloc((size_t)DB * (DE + 3 * DH) * 2);
  float*  h_f     = (float*) alloc((size_t)DB * DH * 4);
  __bf16* h_bf    = (__bf16*)alloc((size_t)DB * DH * 2);
  float*  gx      = (float*) alloc((size_t)DB * 3 * DH * 4);
  float*  gh      = (float*) alloc((size_t)DB * 3 * DH * 4);

  // ---- one-time conversions / precompute ----
  cvt_bf16<<<2048, 256, 0, stream>>>(ehid, ehid_bf, (size_t)DB * DS * 2 * DH);
  cvt_bf16<<<2048, 256, 0, stream>>>(Wih, Wih_bf, (size_t)3 * DH * (DE + 2 * DH));
  cvt_bf16<<<2048, 256, 0, stream>>>(Whh, Whh_bf, (size_t)3 * DH * DH);
  cvt_bf16_t<<<2048, 256, 0, stream>>>(Wk, WkT_bf, 2 * DH, DH);
  cvt_bf16_t<<<2048, 256, 0, stream>>>(Wp, WpT_bf, DE + 3 * DH, DH);
  h0_kernel<<<DB, 256, 0, stream>>>(efin, brW, brb, h_f, h_bf);

  // proj_keys = encoder_hidden @ Wk : M=B*S=8192, N=H=1024, K=2H=2048
  {
    int tiles_n = DH / 64, tiles_total = (DB * DS / 16) * tiles_n;
    wmma_gemm_nt<<<(tiles_total + 7) / 8, 256, 0, stream>>>(
        ehid_bf, 2 * DH, WkT_bf, 2 * DH, pk, DH, nullptr,
        tiles_n, 2 * DH, tiles_total);
  }

  // ---- recurrent steps ----
  for (int t = 0; t < DT; ++t) {
    embed_kernel<<<DB, 128, 0, stream>>>(trg, etbl, xbuf, cat2, t);
    attn_kernel<<<DB, 256, 0, stream>>>(h_f, pk, ehid, mask, xbuf, cat2);

    // gx = x @ W_ih^T : M=64, N=3072, K=2560
    {
      int tiles_n = (3 * DH) / 64, tiles_total = (DB / 16) * tiles_n;
      wmma_gemm_nt<<<(tiles_total + 7) / 8, 256, 0, stream>>>(
          xbuf, DE + 2 * DH, Wih_bf, DE + 2 * DH, gx, 3 * DH, nullptr,
          tiles_n, DE + 2 * DH, tiles_total);
    }
    // gh = h @ W_hh^T : M=64, N=3072, K=1024
    {
      int tiles_n = (3 * DH) / 64, tiles_total = (DB / 16) * tiles_n;
      wmma_gemm_nt<<<(tiles_total + 7) / 8, 256, 0, stream>>>(
          h_bf, DH, Whh_bf, DH, gh, 3 * DH, nullptr,
          tiles_n, DH, tiles_total);
    }
    gate_kernel<<<DB, 256, 0, stream>>>(gx, gh, bih, bhh, h_f, h_bf, cat2);

    // pre = [e_t, h_new, ctx] @ Wp + bp -> out[:, t, :]
    // M=64, N=1024, K=3584 ; C row stride = T*H, base offset = t*H
    {
      int tiles_n = DH / 64, tiles_total = (DB / 16) * tiles_n;
      wmma_gemm_nt<<<(tiles_total + 7) / 8, 256, 0, stream>>>(
          cat2, DE + 3 * DH, WpT_bf, DE + 3 * DH,
          out + (size_t)t * DH, DT * DH, bp,
          tiles_n, DE + 3 * DH, tiles_total);
    }
  }
}
